// BitLinearMLPTriton_77421080477790
// MI455X (gfx1250) — 8-bit matrix operand order repaired and checked by reading, compiled, not run on hardware
//
#include <hip/hip_runtime.h>
#include <hip/hip_bf16.h>
#include <math.h>

typedef __attribute__((ext_vector_type(8))) int v8i;

#define EPSQ 1e-5f

static constexpr int Dd = 2048;
static constexpr int Hh = 8192;
static constexpr int Mtok = 4 * 4096;

__global__ void zero_ws_kernel(float* f) {
  if (threadIdx.x < 2) f[threadIdx.x] = 0.0f;
}

__global__ void abs_sum_kernel(const float* __restrict__ w, size_t n,
                               float* __restrict__ out) {
  __shared__ float red[256];
  size_t stride = (size_t)gridDim.x * blockDim.x;
  float s = 0.0f;
  for (size_t i = (size_t)blockIdx.x * blockDim.x + threadIdx.x; i < n; i += stride)
    s += fabsf(w[i]);
  red[threadIdx.x] = s;
  __syncthreads();
  for (int off = 128; off > 0; off >>= 1) {
    if ((int)threadIdx.x < off) red[threadIdx.x] += red[threadIdx.x + off];
    __syncthreads();
  }
  if (threadIdx.x == 0) atomicAdd(out, red[0]);
}

__global__ void w_quant_kernel(const float* __restrict__ w, size_t n,
                               const float* __restrict__ sum, float inv_n,
                               signed char* __restrict__ q) {
  const float mean  = fmaxf(sum[0] * inv_n, EPSQ);
  const float scale = 1.0f / mean;
  size_t stride = (size_t)gridDim.x * blockDim.x;
  for (size_t i = (size_t)blockIdx.x * blockDim.x + threadIdx.x; i < n; i += stride) {
    float v = rintf(w[i] * scale);
    v = fminf(fmaxf(v, -1.0f), 1.0f);
    q[i] = (signed char)(int)v;
  }
}

__global__ void act_quant_kernel(const float* __restrict__ x, int cols,
                                 signed char* __restrict__ q,
                                 float* __restrict__ dq) {
  __shared__ float red[256];
  const int row = blockIdx.x;
  const float* xr = x + (size_t)row * cols;
  float m = 0.0f;
  for (int c = threadIdx.x; c < cols; c += blockDim.x)
    m = fmaxf(m, fabsf(xr[c]));
  red[threadIdx.x] = m;
  __syncthreads();
  for (int off = 128; off > 0; off >>= 1) {
    if ((int)threadIdx.x < off)
      red[threadIdx.x] = fmaxf(red[threadIdx.x], red[threadIdx.x + off]);
    __syncthreads();
  }
  const float amax  = fmaxf(red[0], EPSQ);
  const float scale = 127.0f / amax;
  if (threadIdx.x == 0) dq[row] = amax * (1.0f / 127.0f);
  signed char* qr = q + (size_t)row * cols;
  for (int c = threadIdx.x; c < cols; c += blockDim.x) {
    float v = rintf(xr[c] * scale);
    v = fminf(fmaxf(v, -128.0f), 127.0f);
    qr[c] = (signed char)(int)v;
  }
}

template <bool GELU>
__global__ __launch_bounds__(256)
void gemm_iu8_kernel(const signed char* __restrict__ A,
                     const float* __restrict__ adq,
                     const signed char* __restrict__ W,
                     const float* __restrict__ wsum, float winv_n,
                     float* __restrict__ out,
                     int N, int K) {
  const int lane = threadIdx.x & 31;
  const int wave = threadIdx.x >> 5;
  const int l15  = lane & 15;
  const int half = (lane >> 4) & 1;
  const int bm   = blockIdx.y * 128 + (wave >> 2) * 64;
  const int bn   = blockIdx.x * 256 + (wave & 3) * 64;

  const float wdq = fmaxf(wsum[0] * winv_n, EPSQ);

  const signed char* abase[4];
#pragma unroll
  for (int i = 0; i < 4; ++i)
    abase[i] = A + (size_t)(bm + i * 16 + l15) * K + half * 16;
  const signed char* bbase[4];
#pragma unroll
  for (int j = 0; j < 4; ++j)
    bbase[j] = W + (size_t)(bn + j * 16 + l15) * K + half * 16;

  v8i c[4][4] = {};

  for (int k = 0; k < K; k += 64) {
    v8i a[4];
#pragma unroll
    for (int i = 0; i < 4; ++i) {
      const signed char* p = abase[i] + k;
      int4 d0 = *(const int4*)(p);
      int4 d1 = *(const int4*)(p + 32);
      a[i] = (v8i){d0.x, d0.y, d0.z, d0.w, d1.x, d1.y, d1.z, d1.w};
    }
    v8i b[4];
#pragma unroll
    for (int j = 0; j < 4; ++j) {
      const signed char* p = bbase[j] + k;
      int4 lo = *(const int4*)(p);
      int4 hi = *(const int4*)(p + 32);
      b[j] = (v8i){lo.x, lo.y, lo.z, lo.w, hi.x, hi.y, hi.z, hi.w};
    }
#pragma unroll
    for (int i = 0; i < 4; ++i)
#pragma unroll
      for (int j = 0; j < 4; ++j)
        c[i][j] = __builtin_amdgcn_wmma_i32_16x16x64_iu8(
             true, a[i],  true, b[j], c[i][j],
             false,  false);
  }

#pragma unroll
  for (int i = 0; i < 4; ++i) {
    const int rowbase = bm + i * 16 + half * 8;
#pragma unroll
    for (int r = 0; r < 8; ++r) {
      const int row = rowbase + r;
      const float rs = adq[row] * wdq;
#pragma unroll
      for (int j = 0; j < 4; ++j) {
        const int col = bn + j * 16 + l15;
        float v = (float)c[i][j][r] * rs;
        if (GELU) v = 0.5f * v * (1.0f + erff(v * 0.70710678118654752f));
        out[(size_t)row * N + col] = v;
      }
    }
  }
}

extern "C" void kernel_launch(void* const* d_in, const int* in_sizes, int n_in,
                              void* d_out, int out_size, void* d_ws, size_t ws_size,
                              hipStream_t stream) {
  (void)in_sizes; (void)n_in; (void)out_size; (void)ws_size;
  const float* x  = (const float*)d_in[0];
  const float* w1 = (const float*)d_in[1];
  const float* w2 = (const float*)d_in[2];
  float* out = (float*)d_out;

  char* ws = (char*)d_ws;
  size_t off = 0;
  auto alloc = [&](size_t bytes) -> char* {
    char* p = ws + off;
    off += (bytes + 255) & ~(size_t)255;
    return p;
  };
  float*       wsf = (float*)alloc(2 * sizeof(float));
  float*       xdq = (float*)alloc((size_t)Mtok * sizeof(float));
  float*       hdq = (float*)alloc((size_t)Mtok * sizeof(float));
  signed char* w1q = (signed char*)alloc((size_t)Hh * Dd);
  signed char* w2q = (signed char*)alloc((size_t)Hh * Dd);
  signed char* xq  = (signed char*)alloc((size_t)Mtok * Dd);
  signed char* hq  = (signed char*)alloc((size_t)Mtok * Hh);
  float*       h   = (float*)alloc((size_t)Mtok * Hh * sizeof(float));

  const size_t nw     = (size_t)Hh * Dd;
  const float  winv_n = 1.0f / (float)((double)Hh * (double)Dd);

  zero_ws_kernel<<<1, 32, 0, stream>>>(wsf);

  abs_sum_kernel<<<1024, 256, 0, stream>>>(w1, nw, wsf + 0);
  abs_sum_kernel<<<1024, 256, 0, stream>>>(w2, nw, wsf + 1);

  w_quant_kernel<<<2048, 256, 0, stream>>>(w1, nw, wsf + 0, winv_n, w1q);
  w_quant_kernel<<<2048, 256, 0, stream>>>(w2, nw, wsf + 1, winv_n, w2q);

  act_quant_kernel<<<Mtok, 256, 0, stream>>>(x, Dd, xq, xdq);

  dim3 g1(Hh / 256, Mtok / 128);
  gemm_iu8_kernel<true><<<g1, 256, 0, stream>>>(xq, xdq, w1q, wsf + 0, winv_n,
                                                h, Hh, Dd);

  act_quant_kernel<<<Mtok, 256, 0, stream>>>(h, Hh, hq, hdq);

  dim3 g2(Dd / 256, Mtok / 128);
  gemm_iu8_kernel<false><<<g2, 256, 0, stream>>>(hq, hdq, w2q, wsf + 1, winv_n,
                                                 out, Dd, Hh);
}
